// RetNetEncoder_80195629351207
// MI455X (gfx1250) — matrix operand order repaired and checked by reading, not compiled, not run on hardware
//
#include <hip/hip_runtime.h>
#include <hip/hip_bf16.h>

// ---------------------------------------------------------------------------
// RetNet encoder for MI455X (gfx1250)
//  - all matmuls via v_wmma_f32_16x16x32_bf16
//  - global->LDS staging via GLOBAL_LOAD_ASYNC_TO_LDS_B128 (ASYNCcnt)
//  - double-buffered LDS pipelines (one barrier per K-step)
//  - A fragments: lane half K-interleave per ISA A-table
//  - B fragments: same lane half K-interleave as A (ISA table order)
// ---------------------------------------------------------------------------

typedef __bf16 bf16;
typedef __attribute__((ext_vector_type(16))) __bf16 v16bf;
typedef __attribute__((ext_vector_type(8)))  __bf16 v8bf;
typedef __attribute__((ext_vector_type(8)))  float  v8f;
typedef int v4i_vs __attribute__((vector_size(16)));   // matches builtin param

#define B_     2
#define S_     2048
#define IN_    512
#define CD_    128
#define HID_   1024
#define HEADS_ 8
#define DH_    128
#define FFN_   4096
#define LAT_   256
#define L_     2
#define MROWS_ (B_ * S_)   // 4096

#if __has_builtin(__builtin_amdgcn_global_load_async_to_lds_b128)
#define HAS_ASYNC_LDS 1
#else
#define HAS_ASYNC_LDS 0
#endif

// 16-byte global -> LDS copy (per-lane addresses).
static __device__ __forceinline__ void copy16_g2l(const bf16* g, bf16* l) {
#if HAS_ASYNC_LDS
    __builtin_amdgcn_global_load_async_to_lds_b128((v4i_vs*)g, (v4i_vs*)l, 0, 0);
#else
    *(v8bf*)l = *(const v8bf*)g;
#endif
}

static __device__ __forceinline__ void async_wait0() {
#if HAS_ASYNC_LDS
#if __has_builtin(__builtin_amdgcn_s_wait_asynccnt)
    __builtin_amdgcn_s_wait_asynccnt(0);
#else
    asm volatile("s_wait_asynccnt 0" ::: "memory");
#endif
#endif
}

// Combine two 16B chunks into a 16-element fragment (keeps 16B alignment).
static __device__ __forceinline__ v16bf frag16(const bf16* p0, const bf16* p1) {
    v8bf lo = *(const v8bf*)p0;
    v8bf hi = *(const v8bf*)p1;
    v16bf r;
#pragma unroll
    for (int i = 0; i < 8; ++i) { r[i] = lo[i]; r[i + 8] = hi[i]; }
    return r;
}

// A-operand fragment: lane half holds K={0..7,16..23} / {8..15,24..31}.
static __device__ __forceinline__ v16bf fragA(const bf16* rowBase, int lane) {
    const bf16* p = rowBase + ((lane < 16) ? 0 : 8);
    return frag16(p, p + 16);
}

// B-operand fragment: lane half holds K={0..7,16..23} / {8..15,24..31}.
static __device__ __forceinline__ v16bf fragB(const bf16* rowBase, int lane) {
    const bf16* p = rowBase + ((lane < 16) ? 0 : 8);
    return frag16(p, p + 16);
}

static __device__ __forceinline__ v8f wmma_bf16(v16bf a, v16bf b, v8f c) {
    return __builtin_amdgcn_wmma_f32_16x16x32_bf16(false, a, false, b, (short)0, c,
                                                   false, false);
}

// ---------------------------------------------------------------------------
// Elementwise helpers
// ---------------------------------------------------------------------------

__global__ __launch_bounds__(256) void cast_f32_bf16_kernel(
    const float* __restrict__ src, bf16* __restrict__ dst, long n) {
    long i = (long)blockIdx.x * 256 + threadIdx.x;
    if (i < n) dst[i] = (bf16)src[i];
}

__global__ __launch_bounds__(256) void concat_cast_kernel(
    const float* __restrict__ x, const float* __restrict__ c,
    bf16* __restrict__ out) {
    long i = (long)blockIdx.x * 256 + threadIdx.x;
    if (i >= (long)MROWS_ * (IN_ + CD_)) return;
    int  col = (int)(i % (IN_ + CD_));
    long row = i / (IN_ + CD_);
    float v = (col < IN_) ? x[row * IN_ + col] : c[row * CD_ + (col - IN_)];
    out[i] = (bf16)v;
}

__global__ __launch_bounds__(256) void softplus_kernel(float* __restrict__ p, int n) {
    int i = blockIdx.x * 256 + threadIdx.x;
    if (i < n) {
        float y  = 0.5f * p[i];
        float sp = (y > 20.f) ? y : log1pf(expf(y));
        p[i] = sp + 0.001f;
    }
}

// Row LayerNorm over HID_ columns, bf16 output.
__global__ __launch_bounds__(256) void ln_kernel(
    const float* __restrict__ x, const float* __restrict__ s,
    const float* __restrict__ bb, bf16* __restrict__ out) {
    __shared__ float red[16];
    const int row = blockIdx.x;
    const int tid = threadIdx.x;
    const float* xr = x + (long)row * HID_;
    float vals[4];
    float lsum = 0.f, lss = 0.f;
#pragma unroll
    for (int i = 0; i < 4; ++i) {
        float xv = xr[tid + i * 256];
        vals[i] = xv;
        lsum += xv;
        lss  += xv * xv;
    }
#pragma unroll
    for (int m = 1; m < 32; m <<= 1) {
        lsum += __shfl_xor(lsum, m, 32);
        lss  += __shfl_xor(lss,  m, 32);
    }
    if ((tid & 31) == 0) { red[tid >> 5] = lsum; red[8 + (tid >> 5)] = lss; }
    __syncthreads();
    float tsum = 0.f, tss = 0.f;
#pragma unroll
    for (int w = 0; w < 8; ++w) { tsum += red[w]; tss += red[8 + w]; }
    float mean = tsum * (1.f / HID_);
    float var  = tss * (1.f / HID_) - mean * mean;
    float rstd = rsqrtf(var + 1e-5f);
#pragma unroll
    for (int i = 0; i < 4; ++i) {
        int c = tid + i * 256;
        out[(long)row * HID_ + c] = (bf16)((vals[i] - mean) * rstd * s[c] + bb[c]);
    }
}

// ---------------------------------------------------------------------------
// Generic bf16 WMMA GEMM:  C[M,N] = A[M,K] @ B[K,N]  (+ epilogue)
// Block: 256 threads (8 waves). Tile 128x128, K-step 32, double-buffered LDS.
// Wave (wm in 0..1, wn in 0..3) computes 64x32 = 4x2 WMMA tiles.
// EPI: 0 none, 1 +bias, 2 +bias+gelu, 3 +bias+residual, 4 +residual
// Requires M%128==0, N%128==0, K%32==0 (true for all call sites here).
// ---------------------------------------------------------------------------

template <int EPI>
__global__ __launch_bounds__(256) void gemm_bf16_wmma(
    const bf16* __restrict__ A, const bf16* __restrict__ Bm,
    float* __restrict__ C, bf16* __restrict__ Cb,
    const float* __restrict__ bias, const float* __restrict__ resid,
    int M, int N, int K) {
    __shared__ __align__(16) bf16 As[2][128][40];   // [m][k], padded rows (80B)
    __shared__ __align__(16) bf16 Bs[2][128][40];   // [n][k] transposed, padded

    const int tid  = threadIdx.x;
    const int lane = tid & 31;
    const int wave = tid >> 5;
    const int wm   = wave >> 2;   // 0..1
    const int wn   = wave & 3;    // 0..3
    const int m0   = blockIdx.y * 128;
    const int n0   = blockIdx.x * 128;

    v8f acc[4][2];
#pragma unroll
    for (int mt = 0; mt < 4; ++mt)
#pragma unroll
        for (int nt = 0; nt < 2; ++nt) acc[mt][nt] = (v8f){};

    const int arow = tid >> 1;          // 0..127
    const int acol = (tid & 1) * 16;    // 0 or 16
    const int bk   = tid & 31;          // 0..31
    const int bn   = (tid >> 5) * 16;   // 0..112

    const int lr = lane & 15;

    auto stage = [&](int buf, int k0) {
        // A tile: contiguous 32B per thread -> async direct-to-LDS
        const bf16* ag = A + (long)(m0 + arow) * K + k0 + acol;
        copy16_g2l(ag,     &As[buf][arow][acol]);
        copy16_g2l(ag + 8, &As[buf][arow][acol + 8]);
        // B tile transposed: Bs[n][k] = B[k0+k][n0+n] (manual, layout change)
        const bf16* bg = Bm + (long)(k0 + bk) * N + n0 + bn;
        v8bf b0 = *(const v8bf*)(bg);
        v8bf b1 = *(const v8bf*)(bg + 8);
#pragma unroll
        for (int i = 0; i < 8; ++i) {
            Bs[buf][bn + i][bk]     = b0[i];
            Bs[buf][bn + 8 + i][bk] = b1[i];
        }
    };

    stage(0, 0);
    async_wait0();
    __syncthreads();

    int cur = 0;
    for (int k0 = 0; k0 < K; k0 += 32) {
        if (k0 + 32 < K) stage(cur ^ 1, k0 + 32);   // prefetch next tile

        v16bf af[4], bfm[2];
#pragma unroll
        for (int mt = 0; mt < 4; ++mt)
            af[mt] = fragA(&As[cur][wm * 64 + mt * 16 + lr][0], lane);
#pragma unroll
        for (int nt = 0; nt < 2; ++nt)
            bfm[nt] = fragB(&Bs[cur][wn * 32 + nt * 16 + lr][0], lane);
#pragma unroll
        for (int mt = 0; mt < 4; ++mt)
#pragma unroll
            for (int nt = 0; nt < 2; ++nt)
                acc[mt][nt] = wmma_bf16(af[mt], bfm[nt], acc[mt][nt]);

        async_wait0();
        __syncthreads();
        cur ^= 1;
    }

    // --- epilogue ---
    const int rhalf = (lane < 16) ? 0 : 8;
#pragma unroll
    for (int mt = 0; mt < 4; ++mt) {
#pragma unroll
        for (int nt = 0; nt < 2; ++nt) {
            const int gcol = n0 + wn * 32 + nt * 16 + lr;
            float bv = 0.f;
            if (EPI == 1 || EPI == 2 || EPI == 3) bv = bias[gcol];
#pragma unroll
            for (int r = 0; r < 8; ++r) {
                const int  grow = m0 + wm * 64 + mt * 16 + r + rhalf;
                const long off  = (long)grow * N + gcol;
                float v = acc[mt][nt][r] + bv;
                if (EPI == 2) {  // tanh-approx gelu (jax default)
                    float x3 = v * v * v;
                    v = 0.5f * v * (1.f + tanhf(0.7978845608f * (v + 0.044715f * x3)));
                }
                if (EPI == 3 || EPI == 4) v += resid[off];
                if (C)  C[off]  = v;
                if (Cb) Cb[off] = (bf16)v;
            }
        }
    }
}

// ---------------------------------------------------------------------------
// Multi-scale retention (parallel form) + per-head groupnorm + silu gating.
// Block = 128 threads (4 waves) handles 64 rows of one (b,h).
// Double-buffered K/V staging; each wave owns 16 rows.
// ---------------------------------------------------------------------------

__global__ __launch_bounds__(128) void retention_kernel(
    const bf16* __restrict__ q, const bf16* __restrict__ k,
    const bf16* __restrict__ v, const float* __restrict__ gpre,
    const float* __restrict__ gns, const float* __restrict__ gnb,
    bf16* __restrict__ go) {
    __shared__ __align__(16) bf16 Kt[2][32][128];     // [m][kd] row-major
    __shared__ __align__(16) bf16 Vt[2][128][40];     // [dh][m] transposed, padded
    __shared__ __align__(16) bf16 Ps[4][16][40];      // per-wave P tile [n][m]

    const int tid  = threadIdx.x;
    const int lane = tid & 31;
    const int wave = tid >> 5;
    const int nb64 = S_ / 64;                         // 32 row-blocks
    const int bh   = blockIdx.x / nb64;
    const int t64  = blockIdx.x % nb64;
    const int b    = bh / HEADS_;
    const int h    = bh % HEADS_;
    const int n0   = t64 * 64 + wave * 16;

    const float log2g = log2f(1.0f - exp2f(-5.0f - (float)h));
    const float scale = 0.08838834764831845f;         // 1/sqrt(128)
    const long  base  = ((long)b * S_) * HID_ + h * DH_;

    const int lr    = lane & 15;
    const int rbase = (lane < 16) ? 0 : 8;

    // Preload Q fragments (A operand) for this wave's 16 rows.
    v16bf qf[4];
#pragma unroll
    for (int kk = 0; kk < 4; ++kk)
        qf[kk] = fragA(q + base + (long)(n0 + lr) * HID_ + kk * 32, lane);

    v8f o[8];
#pragma unroll
    for (int t = 0; t < 8; ++t) o[t] = (v8f){};

    auto stageKV = [&](int buf, int mc) {
        // K chunk [32 rows x 128 kd]: contiguous 64B/thread -> async to LDS
        const int m  = tid >> 2;
        const int qd = (tid & 3) * 32;
        const bf16* src = k + base + (long)(mc + m) * HID_ + qd;
        copy16_g2l(src,      &Kt[buf][m][qd]);
        copy16_g2l(src + 8,  &Kt[buf][m][qd + 8]);
        copy16_g2l(src + 16, &Kt[buf][m][qd + 16]);
        copy16_g2l(src + 24, &Kt[buf][m][qd + 24]);
        // V chunk transposed: Vt[dh][m] (manual, layout change)
#pragma unroll
        for (int it = 0; it < 4; ++it) {
            const int cid = tid + it * 128;   // 0..511
            const int mm  = cid >> 4;         // 0..31
            const int dh0 = (cid & 15) * 8;
            v8bf vv = *(const v8bf*)(v + base + (long)(mc + mm) * HID_ + dh0);
#pragma unroll
            for (int i = 0; i < 8; ++i) Vt[buf][dh0 + i][mm] = vv[i];
        }
    };

    const int m_hi = t64 * 64 + 63;
    stageKV(0, 0);
    async_wait0();
    __syncthreads();

    int cur = 0;
    for (int mc = 0; mc <= m_hi; mc += 32) {
        if (mc + 32 <= m_hi) stageKV(cur ^ 1, mc + 32);

        if (mc <= n0 + 15) {   // wave-uniform guard (EXEC stays all-ones)
            // scores: [16 n] x [32 m], K-dim = 128; preload all K fragments.
            // B operand = K^T: lane j = column m, kd interleaved by lane half.
            v16bf kf[8];
#pragma unroll
            for (int kk = 0; kk < 4; ++kk) {
                kf[2 * kk]     = fragB(&Kt[cur][lr][kk * 32],      lane);
                kf[2 * kk + 1] = fragB(&Kt[cur][16 + lr][kk * 32], lane);
            }
            v8f s0 = (v8f){}, s1 = (v8f){};
#pragma unroll
            for (int kk = 0; kk < 4; ++kk) {
                s0 = wmma_bf16(qf[kk], kf[2 * kk],     s0);
                s1 = wmma_bf16(qf[kk], kf[2 * kk + 1], s1);
            }
            // causal decay weighting, write P (bf16) to wave-private LDS
#pragma unroll
            for (int r = 0; r < 8; ++r) {
                const int n  = n0 + r + rbase;
                const int m1 = mc + lr;
                const int m2 = mc + 16 + lr;
                float d1 = (float)(n - m1);
                float d2 = (float)(n - m2);
                float w1 = (d1 >= 0.f) ? scale * exp2f(log2g * d1) : 0.f;
                float w2 = (d2 >= 0.f) ? scale * exp2f(log2g * d2) : 0.f;
                Ps[wave][r + rbase][lr]      = (bf16)(s0[r] * w1);
                Ps[wave][r + rbase][16 + lr] = (bf16)(s1[r] * w2);
            }
            asm volatile("s_wait_dscnt 0" ::: "memory");
            // O(16x128) += P(16x32) @ V(32x128); preload all V fragments.
            v16bf pf = fragA(&Ps[wave][lr][0], lane);
            v16bf vf[8];
#pragma unroll
            for (int t = 0; t < 8; ++t)
                vf[t] = fragB(&Vt[cur][t * 16 + lr][0], lane);
#pragma unroll
            for (int t = 0; t < 8; ++t) o[t] = wmma_bf16(pf, vf[t], o[t]);
        }

        async_wait0();
        __syncthreads();
        cur ^= 1;
    }

    // Per-row groupnorm over DH=128, scale/shift, silu gate, bf16 store.
#pragma unroll
    for (int r = 0; r < 8; ++r) {
        const int n = n0 + r + rbase;
        float s = 0.f, s2 = 0.f;
#pragma unroll
        for (int t = 0; t < 8; ++t) { float x = o[t][r]; s += x; s2 += x * x; }
#pragma unroll
        for (int msk = 1; msk < 16; msk <<= 1) {   // reduce within 16-lane half
            s  += __shfl_xor(s,  msk, 32);
            s2 += __shfl_xor(s2, msk, 32);
        }
        const float mean = s * (1.f / DH_);
        const float var  = s2 * (1.f / DH_) - mean * mean;
        const float rstd = rsqrtf(var + 1e-5f);
        const long rowoff = (long)(b * S_ + n) * HID_ + h * DH_;
#pragma unroll
        for (int t = 0; t < 8; ++t) {
            const int col = t * 16 + lr;
            float val  = (o[t][r] - mean) * rstd * gns[h * DH_ + col] + gnb[h * DH_ + col];
            float gp   = gpre[rowoff + col];
            float gate = gp / (1.f + expf(-gp));   // silu
            go[rowoff + col] = (bf16)(val * gate);
        }
    }
}

// ---------------------------------------------------------------------------
// Host side
// ---------------------------------------------------------------------------

static void launch_gemm(int epi, hipStream_t st, const bf16* A, const bf16* Bm,
                        float* C, bf16* Cb, const float* bias, const float* resid,
                        int M, int N, int K) {
    dim3 g(N / 128, M / 128), blk(256);
    switch (epi) {
        case 0: gemm_bf16_wmma<0><<<g, blk, 0, st>>>(A, Bm, C, Cb, bias, resid, M, N, K); break;
        case 1: gemm_bf16_wmma<1><<<g, blk, 0, st>>>(A, Bm, C, Cb, bias, resid, M, N, K); break;
        case 2: gemm_bf16_wmma<2><<<g, blk, 0, st>>>(A, Bm, C, Cb, bias, resid, M, N, K); break;
        case 3: gemm_bf16_wmma<3><<<g, blk, 0, st>>>(A, Bm, C, Cb, bias, resid, M, N, K); break;
        default: gemm_bf16_wmma<4><<<g, blk, 0, st>>>(A, Bm, C, Cb, bias, resid, M, N, K); break;
    }
}

static void cast_w(hipStream_t st, const float* src, bf16* dst, long n) {
    long blocks = (n + 255) / 256;
    cast_f32_bf16_kernel<<<(int)blocks, 256, 0, st>>>(src, dst, n);
}

extern "C" void kernel_launch(void* const* d_in, const int* in_sizes, int n_in,
                              void* d_out, int out_size, void* d_ws, size_t ws_size,
                              hipStream_t stream) {
    (void)in_sizes; (void)n_in; (void)out_size; (void)ws_size;
    const float* x     = (const float*)d_in[0];
    const float* c     = (const float*)d_in[1];
    const float* W_in  = (const float*)d_in[2];
    const float* b_in  = (const float*)d_in[3];
    const float* ln1_s = (const float*)d_in[4];
    const float* ln1_b = (const float*)d_in[5];
    const float* Wq    = (const float*)d_in[6];
    const float* Wk    = (const float*)d_in[7];
    const float* Wv    = (const float*)d_in[8];
    const float* Wg    = (const float*)d_in[9];
    const float* Wo    = (const float*)d_in[10];
    const float* gn_s  = (const float*)d_in[11];
    const float* gn_b  = (const float*)d_in[12];
    const float* ln2_s = (const float*)d_in[13];
    const float* ln2_b = (const float*)d_in[14];
    const float* W1    = (const float*)d_in[15];
    const float* b1    = (const float*)d_in[16];
    const float* W2    = (const float*)d_in[17];
    const float* b2    = (const float*)d_in[18];
    const float* lnf_s = (const float*)d_in[19];
    const float* lnf_b = (const float*)d_in[20];
    const float* W_mu  = (const float*)d_in[21];
    const float* b_mu  = (const float*)d_in[22];
    const float* W_lv  = (const float*)d_in[23];
    const float* b_lv  = (const float*)d_in[24];

    float* out_mu = (float*)d_out;
    float* out_lv = out_mu + (long)MROWS_ * LAT_;

    // bump allocator over workspace
    char*  ws  = (char*)d_ws;
    size_t off = 0;
    auto alloc = [&](size_t bytes) -> void* {
        void* p = ws + off;
        off = (off + bytes + 255) & ~(size_t)255;
        return p;
    };

    bf16* Winb = (bf16*)alloc((size_t)(IN_ + CD_) * HID_ * 2);
    bf16* Wqb  = (bf16*)alloc((size_t)L_ * HID_ * HID_ * 2);
    bf16* Wkb  = (bf16*)alloc((size_t)L_ * HID_ * HID_ * 2);
    bf16* Wvb  = (bf16*)alloc((size_t)L_ * HID_ * HID_ * 2);
    bf16* Wgb  = (bf16*)alloc((size_t)L_ * HID_ * HID_ * 2);
    bf16* Wob  = (bf16*)alloc((size_t)L_ * HID_ * HID_ * 2);
    bf16* W1b  = (bf16*)alloc((size_t)L_ * HID_ * FFN_ * 2);
    bf16* W2b  = (bf16*)alloc((size_t)L_ * FFN_ * HID_ * 2);
    bf16* Wmub = (bf16*)alloc((size_t)HID_ * LAT_ * 2);
    bf16* Wlvb = (bf16*)alloc((size_t)HID_ * LAT_ * 2);

    bf16*  xcb  = (bf16*)alloc((size_t)MROWS_ * (IN_ + CD_) * 2);
    float* h    = (float*)alloc((size_t)MROWS_ * HID_ * 4);
    bf16*  xnb  = (bf16*)alloc((size_t)MROWS_ * HID_ * 2);
    bf16*  qb   = (bf16*)alloc((size_t)MROWS_ * HID_ * 2);
    bf16*  kb   = (bf16*)alloc((size_t)MROWS_ * HID_ * 2);
    bf16*  vb   = (bf16*)alloc((size_t)MROWS_ * HID_ * 2);
    float* gpre = (float*)alloc((size_t)MROWS_ * HID_ * 4);
    bf16*  gob  = (bf16*)alloc((size_t)MROWS_ * HID_ * 2);
    bf16*  f1b  = (bf16*)alloc((size_t)MROWS_ * FFN_ * 2);
    bf16*  hfb  = (bf16*)alloc((size_t)MROWS_ * HID_ * 2);

    // --- weight casts (deterministic, every call) ---
    cast_w(stream, W_in, Winb, (long)(IN_ + CD_) * HID_);
    cast_w(stream, Wq,   Wqb,  (long)L_ * HID_ * HID_);
    cast_w(stream, Wk,   Wkb,  (long)L_ * HID_ * HID_);
    cast_w(stream, Wv,   Wvb,  (long)L_ * HID_ * HID_);
    cast_w(stream, Wg,   Wgb,  (long)L_ * HID_ * HID_);
    cast_w(stream, Wo,   Wob,  (long)L_ * HID_ * HID_);
    cast_w(stream, W1,   W1b,  (long)L_ * HID_ * FFN_);
    cast_w(stream, W2,   W2b,  (long)L_ * FFN_ * HID_);
    cast_w(stream, W_mu, Wmub, (long)HID_ * LAT_);
    cast_w(stream, W_lv, Wlvb, (long)HID_ * LAT_);

    // --- input concat + proj: h = [x|c] @ W_in + b_in ---
    {
        long n = (long)MROWS_ * (IN_ + CD_);
        concat_cast_kernel<<<(int)((n + 255) / 256), 256, 0, stream>>>(x, c, xcb);
    }
    launch_gemm(1, stream, xcb, Winb, h, nullptr, b_in, nullptr,
                MROWS_, HID_, IN_ + CD_);

    for (int l = 0; l < L_; ++l) {
        const size_t wq_off = (size_t)l * HID_ * HID_;
        // LN1 -> xn (bf16)
        ln_kernel<<<MROWS_, 256, 0, stream>>>(h, ln1_s + l * HID_, ln1_b + l * HID_, xnb);
        // projections
        launch_gemm(0, stream, xnb, Wqb + wq_off, nullptr, qb, nullptr, nullptr,
                    MROWS_, HID_, HID_);
        launch_gemm(0, stream, xnb, Wkb + wq_off, nullptr, kb, nullptr, nullptr,
                    MROWS_, HID_, HID_);
        launch_gemm(0, stream, xnb, Wvb + wq_off, nullptr, vb, nullptr, nullptr,
                    MROWS_, HID_, HID_);
        launch_gemm(0, stream, xnb, Wgb + wq_off, gpre, nullptr, nullptr, nullptr,
                    MROWS_, HID_, HID_);
        // retention + groupnorm + silu gate -> go (bf16)
        retention_kernel<<<B_ * HEADS_ * (S_ / 64), 128, 0, stream>>>(
            qb, kb, vb, gpre, gn_s + l * HID_, gn_b + l * HID_, gob);
        // h += go @ Wo
        launch_gemm(4, stream, gob, Wob + wq_off, h, nullptr, nullptr, h,
                    MROWS_, HID_, HID_);
        // FFN
        ln_kernel<<<MROWS_, 256, 0, stream>>>(h, ln2_s + l * HID_, ln2_b + l * HID_, xnb);
        launch_gemm(2, stream, xnb, W1b + (size_t)l * HID_ * FFN_, nullptr, f1b,
                    b1 + l * FFN_, nullptr, MROWS_, FFN_, HID_);
        launch_gemm(3, stream, f1b, W2b + (size_t)l * FFN_ * HID_, h, nullptr,
                    b2 + l * HID_, h, MROWS_, HID_, FFN_);
    }

    // final LN + heads
    ln_kernel<<<MROWS_, 256, 0, stream>>>(h, lnf_s, lnf_b, hfb);
    launch_gemm(1, stream, hfb, Wmub, out_mu, nullptr, b_mu, nullptr,
                MROWS_, LAT_, HID_);
    launch_gemm(1, stream, hfb, Wlvb, out_lv, nullptr, b_lv, nullptr,
                MROWS_, LAT_, HID_);
    {
        int n = MROWS_ * LAT_;
        softplus_kernel<<<(n + 255) / 256, 256, 0, stream>>>(out_lv, n);
    }
}
